// SinkhornKnopp_32633161515411
// MI455X (gfx1250) — matrix operand order repaired and checked by reading, not compiled, not run on hardware
//
#include <hip/hip_runtime.h>
#include <hip/hip_fp16.h>

typedef __attribute__((ext_vector_type(16))) _Float16 v16h;
typedef __attribute__((ext_vector_type(8)))  float    v8f;

#define SK_EPS   1e-8f
#define SK_ITERS 20

union V16 { v16h v; int i[8]; };

__device__ __forceinline__ int pack2h(float lo, float hi) {
    __half2 h = __floats2half2_rn(lo, hi);
    return __builtin_bit_cast(int, h);
}

// One wave (32 lanes) processes one 64x64 matrix entirely.
// H0 = exp(M) kept (a) f32 in LDS (stride-68 padded, conflict-free),
// (b) f16 in registers in both WMMA A-layout and B-layout.
// Iterate only the diagonal scale vectors r (row) and c (col):
//   u = H0*c  (WMMA: A=H0_A, B=c-broadcast)  -> r *= rcp(r*u + eps)
//   v = H0^T*r(WMMA: A=r-broadcast, B=H0_B)  -> c *= rcp(c*v + eps)
// Final: out = diag(r) * H0_f32 * diag(c).
__global__ __launch_bounds__(64) void sinkhorn_knopp_wmma(const float* __restrict__ M,
                                                          float* __restrict__ out,
                                                          int nmat) {
    __shared__ float4 lds4[2][64 * 17 + 32];   // per-wave: 64 rows x 68 floats + 128 floats (r|c)

    const int tid = threadIdx.x;
    const int w   = tid >> 5;        // wave in block (2 waves)
    const int l   = tid & 31;        // lane
    const int ml  = l & 15;
    const int p   = l >> 4;          // half select (0: lanes 0-15, 1: lanes 16-31)
    const int mat = blockIdx.x * 2 + w;
    if (mat >= nmat) return;         // wave-uniform

    float* Lf = (float*)&lds4[w][0];          // f32 H0, row stride 68 floats
    float* RC = (float*)&lds4[w][64 * 17];    // [0..63]=r, [64..127]=c

    // ---------------- stage H0 = exp(M) into LDS (f32) ----------------
    const float4* g4 = (const float4*)(M + (size_t)mat * 4096);
    #pragma unroll
    for (int i = 0; i < 32; ++i) {
        float4 x = g4[i * 32 + l];            // elements 128*i + 4*l .. +3
        float4 h;
        h.x = __builtin_amdgcn_exp2f(x.x * 1.4426950408889634f);
        h.y = __builtin_amdgcn_exp2f(x.y * 1.4426950408889634f);
        h.z = __builtin_amdgcn_exp2f(x.z * 1.4426950408889634f);
        h.w = __builtin_amdgcn_exp2f(x.w * 1.4426950408889634f);
        lds4[w][(2 * i + p) * 17 + ml] = h;   // row = 2i+p, cols 4*ml..+3
    }
    __syncthreads();

    // ---------------- build f16 register copies of H0 ----------------
    V16 Ah[4][2];   // A-layout: row-block I (rows 16I..), K-chunk k2 (cols 32*k2..)
    V16 Bh[4][2];   // B-layout: col-block J (cols 16J..), K-chunk k2 (rows 32*k2..)
    #pragma unroll
    for (int I = 0; I < 4; ++I)
        #pragma unroll
        for (int k2 = 0; k2 < 2; ++k2)
            #pragma unroll
            for (int j = 0; j < 8; ++j) {
                int Kloc = 16 * (j >> 2) + 2 * (j & 3) + 8 * p;   // A-layout K for vgpr j, half p
                int row  = 16 * I + ml;
                int col  = 32 * k2 + Kloc;
                Ah[I][k2].i[j] = pack2h(Lf[row * 68 + col], Lf[row * 68 + col + 1]);
            }
    #pragma unroll
    for (int J = 0; J < 4; ++J)
        #pragma unroll
        for (int k2 = 0; k2 < 2; ++k2)
            #pragma unroll
            for (int j = 0; j < 8; ++j) {
                int krow = 32 * k2 + 16 * (j >> 2) + 2 * (j & 3) + 8 * p;   // B-layout: table-order K (matches A-layout)
                int col  = 16 * J + ml;
                Bh[J][k2].i[j] = pack2h(Lf[krow * 68 + col], Lf[(krow + 1) * 68 + col]);
            }

    // ---------------- scale vectors ----------------
    // r: "R-layout": rr[I][j] lanes0-15 = r[16I+j], lanes16-31 = r[16I+8+j]  (matches WMMA D)
    // c: compact lane-indexed: c_lo lane l = c[l], c_hi lane l = c[32+l]
    float rr[4][8];
    #pragma unroll
    for (int I = 0; I < 4; ++I)
        #pragma unroll
        for (int j = 0; j < 8; ++j) rr[I][j] = 1.0f;
    float c_lo = 1.0f, c_hi = 1.0f;

    const int idxrot = ((l + 1) & 31) << 2;   // ds_bpermute byte index: rotate-left-1

    for (int it = 0; it < SK_ITERS; ++it) {
        // ---- row step: u = H0*c ; r *= rcp(r*u + eps) ----
        // pack neighbor pairs of c, then scatter to half-uniform B-layout via bpermute
        int rot0 = __builtin_amdgcn_ds_bpermute(idxrot, __float_as_int(c_lo));
        int rot1 = __builtin_amdgcn_ds_bpermute(idxrot, __float_as_int(c_hi));
        int pk0  = pack2h(c_lo, __int_as_float(rot0));   // lane 2t = half2(c[2t], c[2t+1])
        int pk1  = pack2h(c_hi, __int_as_float(rot1));
        V16 B0, B1;
        #pragma unroll
        for (int j = 0; j < 8; ++j) {
            int idx = (8 * p + 2 * j + 8 * (j >> 2)) << 2;   // table order: lane 8p + 2j (+8 for j>=4)
            B0.i[j] = __builtin_amdgcn_ds_bpermute(idx, pk0);
            B1.i[j] = __builtin_amdgcn_ds_bpermute(idx, pk1);
        }
        #pragma unroll
        for (int I = 0; I < 4; ++I) {
            v8f acc = {};
            acc = __builtin_amdgcn_wmma_f32_16x16x32_f16(false, Ah[I][0].v, false, B0.v,
                                                         (short)0, acc, false, false);
            acc = __builtin_amdgcn_wmma_f32_16x16x32_f16(false, Ah[I][1].v, false, B1.v,
                                                         (short)0, acc, false, false);
            // acc[j] = u[16I+j] (lanes0-15) / u[16I+8+j] (lanes16-31) == rr layout
            #pragma unroll
            for (int j = 0; j < 8; ++j)
                rr[I][j] = rr[I][j] * __builtin_amdgcn_rcpf(fmaf(rr[I][j], acc[j], SK_EPS));
        }

        // ---- col step: v = H0^T * r ; c *= rcp(c*v + eps) ----
        // r-broadcast A operands: one cvt_pk per vgpr, halves line up with A-layout K+8 split
        V16 Ar0, Ar1;
        #pragma unroll
        for (int j = 0; j < 8; ++j) {
            int b = j >> 2, q = 2 * (j & 3);
            Ar0.i[j] = pack2h(rr[b][q],     rr[b][q + 1]);       // K-chunk 0: rows 0..31
            Ar1.i[j] = pack2h(rr[2 + b][q], rr[2 + b][q + 1]);   // K-chunk 1: rows 32..63
        }
        float vv[4];
        #pragma unroll
        for (int J = 0; J < 4; ++J) {
            v8f acc = {};
            acc = __builtin_amdgcn_wmma_f32_16x16x32_f16(false, Ar0.v, false, Bh[J][0].v,
                                                         (short)0, acc, false, false);
            acc = __builtin_amdgcn_wmma_f32_16x16x32_f16(false, Ar1.v, false, Bh[J][1].v,
                                                         (short)0, acc, false, false);
            vv[J] = acc[0];   // lane-indexed: v[16J + (l&15)] (every row of D identical)
        }
        float vlo = p ? vv[1] : vv[0];   // lane l -> v[l]
        float vhi = p ? vv[3] : vv[2];   // lane l -> v[32+l]
        c_lo = c_lo * __builtin_amdgcn_rcpf(fmaf(c_lo, vlo, SK_EPS));
        c_hi = c_hi * __builtin_amdgcn_rcpf(fmaf(c_hi, vhi, SK_EPS));
    }

    // ---------------- finale: out = diag(r) * H0_f32 * diag(c) ----------------
    #pragma unroll
    for (int I = 0; I < 4; ++I)
        #pragma unroll
        for (int j = 0; j < 8; ++j)
            RC[16 * I + 8 * p + j] = rr[I][j];   // all 16 lanes of a half write same value
    RC[64 + l]      = c_lo;
    RC[96 + l]      = c_hi;
    __syncthreads();

    float4 cvec = *(const float4*)&RC[64 + 4 * ml];   // c[4*ml .. +3]
    float4* o4  = (float4*)(out + (size_t)mat * 4096);
    #pragma unroll
    for (int i = 0; i < 32; ++i) {
        int   row = 2 * i + p;
        float rm  = RC[row];
        float4 h  = lds4[w][row * 17 + ml];
        float4 ov;
        ov.x = h.x * (rm * cvec.x);
        ov.y = h.y * (rm * cvec.y);
        ov.z = h.z * (rm * cvec.z);
        ov.w = h.w * (rm * cvec.w);
        o4[i * 32 + l] = ov;
    }
}

extern "C" void kernel_launch(void* const* d_in, const int* in_sizes, int n_in,
                              void* d_out, int out_size, void* d_ws, size_t ws_size,
                              hipStream_t stream) {
    const float* M = (const float*)d_in[0];
    float* out = (float*)d_out;
    const int nmat = in_sizes[0] / 4096;           // 16384 matrices of 64x64
    dim3 block(64);                                 // 2 waves, 1 matrix per wave
    dim3 grid((nmat + 1) / 2);
    sinkhorn_knopp_wmma<<<grid, block, 0, stream>>>(M, out, nmat);
}
